// LaminiIndex_77498389889796
// MI455X (gfx1250) — matrix operand order repaired and checked by reading, not compiled, not run on hardware
//
#include <hip/hip_runtime.h>
#include <hip/hip_bf16.h>

// Problem constants (from reference setup_inputs)
#define M_TOT   2048      // B*L = 4*512
#define N_TOT   65536
#define D_DIM   384
#define TOPK    8
#define ROWS    16        // M-tile per block (one WMMA M)
#define NCHUNK  1024      // N columns scored per block (64 KB LDS of f32 scores)
#define CHUNKS  (N_TOT / NCHUNK)   // 64
#define KSTEPS  (D_DIM / 32)       // 12 wmma K-steps

typedef __attribute__((ext_vector_type(16))) _Float16 v16h;
typedef __attribute__((ext_vector_type(8)))  _Float16 v8h;
typedef __attribute__((ext_vector_type(4)))  _Float16 v4h;
typedef __attribute__((ext_vector_type(8)))  float    v8f;

#define NEG_INF (-__builtin_inff())

// ---------------------------------------------------------------------------
// Sorted descending top-8 insert, fully static indexing (stays in VGPRs).
// ---------------------------------------------------------------------------
__device__ __forceinline__ void topk_insert(float (&tv)[8], int (&ti)[8], float v, int idx) {
    if (v <= tv[7]) return;
    float cv = v; int ci = idx;
#pragma unroll
    for (int s = 0; s < 8; ++s) {
        bool gt = cv > tv[s];
        float nv = gt ? tv[s] : cv;
        int   ni = gt ? ti[s] : ci;
        tv[s] = gt ? cv : tv[s];
        ti[s] = gt ? ci : ti[s];
        cv = nv; ci = ni;
    }
}

// ---------------------------------------------------------------------------
// Kernel 0: f32 -> f16 convert (x4 vectorized)
// ---------------------------------------------------------------------------
__global__ __launch_bounds__(256) void convert_f16(const float* __restrict__ src,
                                                   _Float16* __restrict__ dst, int n4) {
    int i = blockIdx.x * blockDim.x + threadIdx.x;
    if (i < n4) {
        float4 v = ((const float4*)src)[i];
        v4h h;
        h[0] = (_Float16)v.x; h[1] = (_Float16)v.y;
        h[2] = (_Float16)v.z; h[3] = (_Float16)v.w;
        ((v4h*)dst)[i] = h;
    }
}

// ---------------------------------------------------------------------------
// Kernel 1: WMMA logits + gumbel, per-chunk top-8 candidates.
// grid = (M_TOT/ROWS, CHUNKS), block = 256 (8 waves)
// ---------------------------------------------------------------------------
__global__ __launch_bounds__(256) void score_topk(
    const _Float16* __restrict__ qh, const _Float16* __restrict__ kh,
    const float* __restrict__ gumbel,
    float* __restrict__ cand_val, int* __restrict__ cand_idx)
{
    __shared__ float s_scores[ROWS * NCHUNK];   // 64 KB

    const int mTile = blockIdx.x;
    const int chunk = blockIdx.y;
    const int nBase = chunk * NCHUNK;
    const int wave  = threadIdx.x >> 5;
    const int lane  = threadIdx.x & 31;
    const int m0    = mTile * ROWS;

    // --- A fragments (16x32 f16 per K-step), resident for the whole chunk ---
    // lane 0-15: M=lane, K = 0..15 ; lane 16-31: M=lane-16, K = 16..31
    const int aRow = m0 + (lane & 15);
    const int aSel = (lane & 16) ? 16 : 0;
    const _Float16* qp = qh + (size_t)aRow * D_DIM + aSel;
    v16h A[KSTEPS];
#pragma unroll
    for (int t = 0; t < KSTEPS; ++t) {
        v8h lo = *(const v8h*)(qp + t * 32);
        v8h hi = *(const v8h*)(qp + t * 32 + 8);
#pragma unroll
        for (int e = 0; e < 8; ++e) { A[t][e] = lo[e]; A[t][8 + e] = hi[e]; }
    }

    // --- B fragments: lane 0-15 -> col n=lane, K=0..15 ; lane 16-31 -> K=16..31 ---
    const int bSel = (lane & 16) ? 16 : 0;
    const int rOff = (lane & 16) ? 8 : 0;

    for (int tile = wave; tile < NCHUNK / 16; tile += 8) {
        const int nCol = nBase + tile * 16 + (lane & 15);
        const _Float16* kp = kh + (size_t)nCol * D_DIM + bSel;
        v8f C = {};
#pragma unroll
        for (int t = 0; t < KSTEPS; ++t) {
            v16h B = *(const v16h*)(kp + t * 32);
            C = __builtin_amdgcn_wmma_f32_16x16x32_f16(
                    false, A[t], false, B, (short)0, C, false, false);
        }
        // C element e -> row (e+rOff), col (tile*16 + lane&15). Add gumbel (NT stream).
        const int cLoc = tile * 16 + (lane & 15);
#pragma unroll
        for (int e = 0; e < 8; ++e) {
            const int r = e + rOff;
            float g = __builtin_nontemporal_load(
                gumbel + (size_t)(m0 + r) * N_TOT + (size_t)(nBase + cLoc));
            s_scores[r * NCHUNK + cLoc] = C[e] + g;
        }
    }
    __syncthreads();

    // --- wave-parallel top-8 per row; wave w owns rows 2w, 2w+1 ---
#pragma unroll
    for (int rr = 0; rr < 2; ++rr) {
        const int r = wave * 2 + rr;
        float tv[8]; int ti[8];
#pragma unroll
        for (int s = 0; s < 8; ++s) { tv[s] = NEG_INF; ti[s] = -1; }
        for (int j = lane; j < NCHUNK; j += 32)
            topk_insert(tv, ti, s_scores[r * NCHUNK + j], nBase + j);

        float wv = 0.0f; int wi = -1;
#pragma unroll
        for (int it = 0; it < TOPK; ++it) {
            float bv = tv[0]; int bi = ti[0];
#pragma unroll
            for (int off = 16; off >= 1; off >>= 1) {
                float ov = __shfl_xor(bv, off, 32);
                int   oi = __shfl_xor(bi, off, 32);
                if (ov > bv || (ov == bv && oi > bi)) { bv = ov; bi = oi; }
            }
            if (ti[0] == bi) {              // unique owner pops its head
#pragma unroll
                for (int s = 0; s < 7; ++s) { tv[s] = tv[s + 1]; ti[s] = ti[s + 1]; }
                tv[7] = NEG_INF; ti[7] = -1;
            }
            if (lane == it) { wv = bv; wi = bi; }
        }
        if (lane < TOPK) {
            size_t base = ((size_t)(m0 + r) * CHUNKS + chunk) * TOPK + lane;
            cand_val[base] = wv;
            cand_idx[base] = wi;
        }
    }
}

// ---------------------------------------------------------------------------
// Kernel 2: merge per-chunk candidates -> global top-8, gather mean of rows.
// grid = M_TOT/8 blocks, 8 waves/block, one wave per query row.
// ---------------------------------------------------------------------------
__global__ __launch_bounds__(256) void merge_gather(
    const float* __restrict__ cand_val, const int* __restrict__ cand_idx,
    const float* __restrict__ keys, const float* __restrict__ values,
    float* __restrict__ out)
{
    const int wave = threadIdx.x >> 5;
    const int lane = threadIdx.x & 31;
    const int m = blockIdx.x * 8 + wave;
    const int NC = CHUNKS * TOPK;   // 512 candidates per row

    float tv[8]; int ti[8];
#pragma unroll
    for (int s = 0; s < 8; ++s) { tv[s] = NEG_INF; ti[s] = -1; }
    const float* cvp = cand_val + (size_t)m * NC;
    const int*   cip = cand_idx + (size_t)m * NC;
    for (int j = lane; j < NC; j += 32)
        topk_insert(tv, ti, cvp[j], cip[j]);

    int fidx[8];
#pragma unroll
    for (int it = 0; it < TOPK; ++it) {
        float bv = tv[0]; int bi = ti[0];
#pragma unroll
        for (int off = 16; off >= 1; off >>= 1) {
            float ov = __shfl_xor(bv, off, 32);
            int   oi = __shfl_xor(bi, off, 32);
            if (ov > bv || (ov == bv && oi > bi)) { bv = ov; bi = oi; }
        }
        if (ti[0] == bi) {
#pragma unroll
            for (int s = 0; s < 7; ++s) { tv[s] = tv[s + 1]; ti[s] = ti[s + 1]; }
            tv[7] = NEG_INF; ti[7] = -1;
        }
        fidx[it] = bi;
    }

    // out = [key_vec (M,D) | value_vec (M,D)], mean of the 8 selected rows
    for (int d = lane; d < D_DIM; d += 32) {
        float aK = 0.f, aV = 0.f;
#pragma unroll
        for (int j = 0; j < TOPK; ++j) {
            size_t off = (size_t)fidx[j] * D_DIM + d;
            aK += keys[off];
            aV += values[off];
        }
        out[(size_t)m * D_DIM + d]               = 0.125f * aK;
        out[(size_t)(M_TOT + m) * D_DIM + d]     = 0.125f * aV;
    }
}

// ---------------------------------------------------------------------------
extern "C" void kernel_launch(void* const* d_in, const int* in_sizes, int n_in,
                              void* d_out, int out_size, void* d_ws, size_t ws_size,
                              hipStream_t stream) {
    const float* query  = (const float*)d_in[0];   // (4,512,384)
    const float* keys   = (const float*)d_in[1];   // (65536,384)
    const float* values = (const float*)d_in[2];   // (65536,384)
    const float* gumbel = (const float*)d_in[3];   // (2048,65536)
    float* out = (float*)d_out;

    const size_t QH_BYTES   = (size_t)M_TOT * D_DIM * sizeof(_Float16);   // 1.5 MB
    const size_t KH_BYTES   = (size_t)N_TOT * D_DIM * sizeof(_Float16);   // 48 MB
    const size_t CAND_BYTES = (size_t)M_TOT * CHUNKS * TOPK * sizeof(float); // 4 MB

    char* ws = (char*)d_ws;
    _Float16* qh       = (_Float16*)ws;
    _Float16* kh       = (_Float16*)(ws + QH_BYTES);
    float*    cand_val = (float*)   (ws + QH_BYTES + KH_BYTES);
    int*      cand_idx = (int*)     (ws + QH_BYTES + KH_BYTES + CAND_BYTES);

    // 1) convert q and keys to f16 (keys f16 = 48 MB -> stays L2-resident)
    {
        int n4q = M_TOT * D_DIM / 4;
        convert_f16<<<(n4q + 255) / 256, 256, 0, stream>>>(query, qh, n4q);
        int n4k = N_TOT * D_DIM / 4;
        convert_f16<<<(n4k + 255) / 256, 256, 0, stream>>>(keys, kh, n4k);
    }

    // 2) WMMA scores + gumbel + per-chunk top-8
    {
        dim3 grid(M_TOT / ROWS, CHUNKS);   // 128 x 64
        score_topk<<<grid, 256, 0, stream>>>(qh, kh, gumbel, cand_val, cand_idx);
    }

    // 3) merge candidates, gather means
    merge_gather<<<M_TOT / 8, 256, 0, stream>>>(cand_val, cand_idx, keys, values, out);
}
